// SelfAttention_20186346291209
// MI455X (gfx1250) — matrix operand order repaired and checked by reading, not compiled, not run on hardware
//
#include <hip/hip_runtime.h>

// ---------------------------------------------------------------------------
// CDNA5 (gfx1250, wave32) block-sparse self-attention, bf16 WMMA everywhere,
// TDM (tensor_load_to_lds) staging of K tiles in the attention kernel.
// ---------------------------------------------------------------------------

typedef __attribute__((ext_vector_type(16))) __bf16 v16bf;
typedef __attribute__((ext_vector_type(8)))  __bf16 bh8;
typedef __attribute__((ext_vector_type(8)))  float  v8f;
typedef __attribute__((ext_vector_type(4)))  unsigned short us4;
typedef __attribute__((ext_vector_type(4)))  unsigned int u32x4;
typedef __attribute__((ext_vector_type(8)))  int i32x8;
typedef __attribute__((ext_vector_type(4)))  int i32x4;

#define HEADS   8
#define HIDDEN  64
#define EMB     512
#define HH      512
#define NTOK    2048
#define BS      4
#define ROWS    (BS * NTOK)   // 8192
#define LOCALC  64
#define NEGBIG  (-1.0e30f)

#ifndef __has_builtin
#define __has_builtin(x) 0
#endif
#if __has_builtin(__builtin_amdgcn_tensor_load_to_lds) && \
    __has_builtin(__builtin_amdgcn_s_wait_tensorcnt)
#define HAVE_TDM 1
#else
#define HAVE_TDM 0
#endif

static __device__ __forceinline__ unsigned short f2bf(float f) {
  __bf16 h = (__bf16)f;
  return __builtin_bit_cast(unsigned short, h);
}
static __device__ __forceinline__ bh8 ldg8(const unsigned short* p) {
  return *(const bh8*)p;
}
static __device__ __forceinline__ v16bf cat16(bh8 lo, bh8 hi) {
  v16bf r;
#pragma unroll
  for (int i = 0; i < 8; ++i) { r[i] = lo[i]; r[i + 8] = hi[i]; }
  return r;
}
static __device__ __forceinline__ v8f vzero8() {
  v8f z = {0.f, 0.f, 0.f, 0.f, 0.f, 0.f, 0.f, 0.f};
  return z;
}

#if HAVE_TDM
// Issue a TDM load of a 16-row x 64-element bf16 tile (row stride 512 elems)
// from global memory into LDS. 2-D descriptor per CDNA5 ISA.
// This toolchain's builtin takes 6 args:
//   (u32x4 group0, i32x8 group1, i32x4 group2, i32x4 group3, i32x8 extra, i32 cpol)
static __device__ __forceinline__ void tdm_load_tile(const unsigned short* gsrc,
                                                     unsigned short* lds_dst) {
  unsigned long long ga = (unsigned long long)(size_t)gsrc;
  u32x4 g0;
  g0[0] = 1u;                                    // count=1, user descriptor
  g0[1] = (unsigned int)(size_t)lds_dst;         // LDS byte address (low 32 bits)
  g0[2] = (unsigned int)(ga & 0xffffffffu);      // global_addr[31:0]
  g0[3] = (unsigned int)((ga >> 32) & 0x01ffffffu) | 0x80000000u; // addr hi | type=2
  i32x8 g1;
  g1[0] = 0x10000;                // data_size = 1 -> 2 bytes
  g1[1] = (int)(512u << 16);      // tensor_dim0 = 512 (bits 63:48)
  g1[2] = (int)(0x2000u << 16);   // tensor_dim1 = 8192 (low 16 bits at 95:80)
  g1[3] = (int)(64u << 16);       // tile_dim0 = 64 (bits 127:112)
  g1[4] = 16;                     // tile_dim1 = 16 (bits 143:128)
  g1[5] = 512;                    // tensor_dim0_stride = 512 (bits 191:160)
  g1[6] = 0;
  g1[7] = 0;
  i32x4 z4 = {0, 0, 0, 0};
  i32x8 z8 = {0, 0, 0, 0, 0, 0, 0, 0};
  __builtin_amdgcn_tensor_load_to_lds(g0, g1, z4, z4, z8, 0);
}
#endif

// ---------------------------------------------------------------------------
// Kernel 1: convert x (fp32) -> bf16, 4 elements / thread.
// ---------------------------------------------------------------------------
__global__ void __launch_bounds__(256) k_cvt_x(const float4* __restrict__ x,
                                               unsigned short* __restrict__ xb) {
  int i = blockIdx.x * blockDim.x + threadIdx.x;   // 1,048,576 threads exactly
  float4 v = x[i];
  us4 u;
  u.x = f2bf(v.x); u.y = f2bf(v.y); u.z = f2bf(v.z); u.w = f2bf(v.w);
  *(us4*)(xb + 4 * (size_t)i) = u;
}

// ---------------------------------------------------------------------------
// Kernel 2: transpose + convert the four 512x512 weight matrices to bf16.
// Output row = output-column of original (so WMMA B lanes read contiguous K).
// Order in wt: [wq_t, wk_t, wv_t, wu_t].
// ---------------------------------------------------------------------------
__global__ void __launch_bounds__(256) k_tw(const float* __restrict__ wq,
                                            const float* __restrict__ wk,
                                            const float* __restrict__ wv,
                                            const float* __restrict__ wu,
                                            unsigned short* __restrict__ wt) {
  int m = blockIdx.y;
  const float* s = (m == 0) ? wq : (m == 1) ? wk : (m == 2) ? wv : wu;
  unsigned short* d = wt + (size_t)m * EMB * HH;
  int idx = blockIdx.x * blockDim.x + threadIdx.x;  // 262,144 per matrix
  int c = idx >> 9, k = idx & 511;
  d[idx] = f2bf(s[k * HH + c]);
}

// ---------------------------------------------------------------------------
// Kernel 3: fused QKV projection. One wave computes one 16x16 output tile.
// z = 0: Q (scaled 0.125, layout [b,t,h,d]); z = 1: K (same); z = 2: V,
// stored transposed as vt[b][h*64+d][t] for the P*V WMMA B layout.
// ---------------------------------------------------------------------------
__global__ void __launch_bounds__(128) k_qkv(const unsigned short* __restrict__ xb,
                                             const unsigned short* __restrict__ wt,
                                             unsigned short* __restrict__ q_ws,
                                             unsigned short* __restrict__ k_ws,
                                             unsigned short* __restrict__ vt_ws) {
  const int wave = threadIdx.x >> 5;
  const int lane = threadIdx.x & 31;
  const int laneq = lane & 15;
  const int z = blockIdx.z;
  const int m0 = blockIdx.x * 16;
  const int n0 = (blockIdx.y * 4 + wave) * 16;

  const unsigned short* BT = wt + (size_t)z * EMB * HH;
  const int rowA = m0 + laneq;
  const int aSel = (lane < 16) ? 0 : 16;   // A: K chunk  {aSel .. aSel+15}
  const int colB = n0 + laneq;
  const int bSel = (lane < 16) ? 0 : 16;   // B: K chunk  {bSel .. bSel+15}
  const unsigned short* ap = xb + (size_t)rowA * EMB;
  const unsigned short* bp = BT + (size_t)colB * EMB;

  v8f c = vzero8();
#pragma unroll 4
  for (int k0 = 0; k0 < EMB; k0 += 32) {
    v16bf av = cat16(ldg8(ap + k0 + aSel), ldg8(ap + k0 + aSel + 8));
    v16bf bv = cat16(ldg8(bp + k0 + bSel), ldg8(bp + k0 + bSel + 8));
    c = __builtin_amdgcn_wmma_f32_16x16x32_bf16(false, av, false, bv,
                                                (short)0, c, false, false);
  }

  if (z < 2) {
    unsigned short* O = (z == 0) ? q_ws : k_ws;
    const int row0 = m0 + ((lane < 16) ? 0 : 8);
#pragma unroll
    for (int r = 0; r < 8; ++r)
      O[(size_t)(row0 + r) * HH + colB] = f2bf(c[r] * 0.125f);
  } else {
    const int b  = m0 >> 11;                       // m0 / 2048
    const int t0 = (m0 & 2047) + ((lane < 16) ? 0 : 8);
    bh8 pk;
#pragma unroll
    for (int r = 0; r < 8; ++r) pk[r] = (__bf16)c[r];
    *(bh8*)(vt_ws + ((size_t)(b * HH + colB)) * NTOK + t0) = pk;
  }
}

// ---------------------------------------------------------------------------
// Kernel 4: block-sparse flash attention. One wave owns one (b, h, 16-row
// query tile). KV blocks: strided residue blocks + 5-block local window.
// K tiles are staged into LDS via the Tensor Data Mover (TENSORcnt-tracked);
// P transposes C-layout -> A-layout through LDS.
// ---------------------------------------------------------------------------
__global__ void __launch_bounds__(128) k_attn(const unsigned short* __restrict__ q_ws,
                                              const unsigned short* __restrict__ k_ws,
                                              const unsigned short* __restrict__ vt_ws,
                                              unsigned short* __restrict__ attn) {
  __shared__ __attribute__((aligned(16))) unsigned short ptile[4][16][32]; // 4 KB
  __shared__ __attribute__((aligned(16))) unsigned short ktile[4][2][16][64]; // 16 KB

  const int wave = threadIdx.x >> 5;
  const int lane = threadIdx.x & 31;
  const int laneq = lane & 15;
  const int half8 = (lane < 16) ? 0 : 8;
  const int asel = (lane < 16) ? 0 : 16;        // A-frag K chunk select
  const int qt = blockIdx.x * 4 + wave;         // 0..127
  const int h = blockIdx.y, b = blockIdx.z;
  const int i0 = qt * 16;

  // Q A-fragments (16x32 each): rows = tokens i0..i0+15, K = d.
  const unsigned short* qp =
      q_ws + ((size_t)(b * NTOK + i0 + laneq)) * HH + h * HIDDEN;
  const v16bf qa0 = cat16(ldg8(qp + asel),      ldg8(qp + asel + 8));
  const v16bf qa1 = cat16(ldg8(qp + 32 + asel), ldg8(qp + 32 + asel + 8));

  v8f o0 = vzero8(), o1 = vzero8(), o2 = vzero8(), o3 = vzero8();
  float mrun[8], lrun[8];
#pragma unroll
  for (int r = 0; r < 8; ++r) { mrun[r] = NEGBIG; lrun[r] = 0.f; }

  const int r0 = i0 & 63;
  int ns = (i0 >> 6) - 1; if (ns < 0) ns = 0;       // strided blocks
  int blmin = qt - 4; if (blmin < 0) blmin = 0;     // local window blocks
  const int L = ns + (qt - blmin + 1);
  const int ngroups = (L + 1) >> 1;

  const int ksel = (lane < 16) ? 0 : 16;   // K B-frag d-chunk select

  // S tile from an LDS-staged K block: B = K^T, K-dim = d (2 x 32).
  auto calc_s_lds = [&](int buf) -> v8f {
    const unsigned short* kp = &ktile[wave][buf][laneq][ksel];
    v16bf kb0 = cat16(*(const bh8*)kp,        *(const bh8*)(kp + 8));
    v16bf kb1 = cat16(*(const bh8*)(kp + 32), *(const bh8*)(kp + 40));
    v8f s = vzero8();
    s = __builtin_amdgcn_wmma_f32_16x16x32_bf16(false, qa0, false, kb0,
                                                (short)0, s, false, false);
    s = __builtin_amdgcn_wmma_f32_16x16x32_bf16(false, qa1, false, kb1,
                                                (short)0, s, false, false);
    return s;
  };

  for (int g = 0; g < ngroups; ++g) {
    const int ta = 2 * g, tb = 2 * g + 1;
    const int j0a = (ta < ns) ? (r0 + 64 * ta) : ((blmin + ta - ns) * 16);
    const bool haveB = (tb < L);
    const int j0b = haveB ? ((tb < ns) ? (r0 + 64 * tb) : ((blmin + tb - ns) * 16))
                          : j0a;

    const unsigned short* srcA =
        k_ws + ((size_t)(b * NTOK + j0a)) * HH + h * HIDDEN;
    const unsigned short* srcB =
        k_ws + ((size_t)(b * NTOK + j0b)) * HH + h * HIDDEN;

    // Ensure previous group's ds reads of ktile have drained before overwrite.
    asm volatile("s_wait_dscnt 0x0" ::: "memory");
#if HAVE_TDM
    tdm_load_tile(srcA, &ktile[wave][0][0][0]);
    if (haveB) tdm_load_tile(srcB, &ktile[wave][1][0][0]);
    __builtin_amdgcn_s_wait_tensorcnt(0);
    asm volatile("" ::: "memory");
#else
    {
      // Fallback: cooperative global -> LDS copy (2 rows per lane-pair).
      const int crow = lane >> 1;
      const int chalf = (lane & 1) * 32;
#pragma unroll
      for (int c = 0; c < 32; c += 8) {
        *(bh8*)&ktile[wave][0][crow][chalf + c] =
            ldg8(srcA + (size_t)crow * HH + chalf + c);
        *(bh8*)&ktile[wave][1][crow][chalf + c] =
            ldg8(srcB + (size_t)crow * HH + chalf + c);
      }
      asm volatile("s_wait_dscnt 0x0" ::: "memory");
    }
#endif

    v8f s0 = calc_s_lds(0);
    v8f s1 = haveB ? calc_s_lds(1) : vzero8();

#pragma unroll
    for (int r = 0; r < 8; ++r) {
      const int row = i0 + half8 + r;
      const int da = row - (j0a + laneq);
      const bool oka = (da >= 0) && ((da <= LOCALC) || ((da & 63) == 0));
      float s0v = oka ? s0[r] : NEGBIG;
      float s1v = NEGBIG;
      if (haveB) {
        const int db = row - (j0b + laneq);
        const bool okb = (db >= 0) && ((db <= LOCALC) || ((db & 63) == 0));
        s1v = okb ? s1[r] : NEGBIG;
      }
      // Row-wise max over 16 lanes (stays inside each 16-lane half).
      float mr = fmaxf(s0v, s1v);
#pragma unroll
      for (int ms = 1; ms < 16; ms <<= 1) mr = fmaxf(mr, __shfl_xor(mr, ms, 32));
      const float mnew  = fmaxf(mrun[r], mr);
      const float alpha = __expf(mrun[r] - mnew);
      const float p0 = __expf(s0v - mnew);
      const float p1 = __expf(s1v - mnew);
      float rs = p0 + p1;
#pragma unroll
      for (int ms = 1; ms < 16; ms <<= 1) rs += __shfl_xor(rs, ms, 32);
      lrun[r] = lrun[r] * alpha + rs;
      mrun[r] = mnew;
      o0[r] *= alpha; o1[r] *= alpha; o2[r] *= alpha; o3[r] *= alpha;
      // P (C layout) -> LDS, row-major bf16 16x32 tile.
      ptile[wave][half8 + r][laneq]      = f2bf(p0);
      ptile[wave][half8 + r][laneq + 16] = f2bf(p1);
    }

    // Same-wave LDS ordering fence (LDS pipe is in-order per wave).
    asm volatile("s_wait_dscnt 0x0" ::: "memory");

    // Re-read P in A layout: row = laneq, K chunk {asel .. asel+15}.
    bh8 plo = *(const bh8*)&ptile[wave][laneq][asel];
    bh8 phi = *(const bh8*)&ptile[wave][laneq][asel + 8];
    v16bf pa = cat16(plo, phi);

    // V B-fragments: K = 32 kv tokens (block A -> lanes 0-15, block B -> 16-31),
    // N = 16 d-columns per WMMA; vt is [b][h*64+d][t].
    const int jb = (lane < 16) ? j0a : j0b;
    const unsigned short* vb_base =
        vt_ws + ((size_t)(b * HH + h * HIDDEN)) * NTOK + jb;
    {
      const unsigned short* vp = vb_base + (size_t)laneq * NTOK;
      v16bf vv = cat16(ldg8(vp), ldg8(vp + 8));
      o0 = __builtin_amdgcn_wmma_f32_16x16x32_bf16(false, pa, false, vv,
                                                   (short)0, o0, false, false);
      vp = vb_base + (size_t)(16 + laneq) * NTOK;
      vv = cat16(ldg8(vp), ldg8(vp + 8));
      o1 = __builtin_amdgcn_wmma_f32_16x16x32_bf16(false, pa, false, vv,
                                                   (short)0, o1, false, false);
      vp = vb_base + (size_t)(32 + laneq) * NTOK;
      vv = cat16(ldg8(vp), ldg8(vp + 8));
      o2 = __builtin_amdgcn_wmma_f32_16x16x32_bf16(false, pa, false, vv,
                                                   (short)0, o2, false, false);
      vp = vb_base + (size_t)(48 + laneq) * NTOK;
      vv = cat16(ldg8(vp), ldg8(vp + 8));
      o3 = __builtin_amdgcn_wmma_f32_16x16x32_bf16(false, pa, false, vv,
                                                   (short)0, o3, false, false);
    }
  }

  // Normalize and store attention output, bf16 [b, t, h*64+d].
#pragma unroll
  for (int r = 0; r < 8; ++r) {
    const float inv = 1.0f / lrun[r];
    const size_t off =
        (size_t)(b * NTOK + i0 + half8 + r) * HH + h * HIDDEN + laneq;
    attn[off]      = f2bf(o0[r] * inv);
    attn[off + 16] = f2bf(o1[r] * inv);
    attn[off + 32] = f2bf(o2[r] * inv);
    attn[off + 48] = f2bf(o3[r] * inv);
  }
}

// ---------------------------------------------------------------------------
// Kernel 5: unify GEMM, out = attn @ w_unify + b_unify, f32 output.
// ---------------------------------------------------------------------------
__global__ void __launch_bounds__(128) k_unify(const unsigned short* __restrict__ attn,
                                               const unsigned short* __restrict__ wut,
                                               const float* __restrict__ bias,
                                               float* __restrict__ out) {
  const int wave = threadIdx.x >> 5;
  const int lane = threadIdx.x & 31;
  const int laneq = lane & 15;
  const int m0 = blockIdx.x * 16;
  const int n0 = (blockIdx.y * 4 + wave) * 16;

  const int rowA = m0 + laneq;
  const int aSel = (lane < 16) ? 0 : 16;
  const int colB = n0 + laneq;
  const int bSel = (lane < 16) ? 0 : 16;
  const unsigned short* ap = attn + (size_t)rowA * HH;
  const unsigned short* bp = wut + (size_t)colB * HH;

  v8f c = vzero8();
#pragma unroll 4
  for (int k0 = 0; k0 < HH; k0 += 32) {
    v16bf av = cat16(ldg8(ap + k0 + aSel), ldg8(ap + k0 + aSel + 8));
    v16bf bv = cat16(ldg8(bp + k0 + bSel), ldg8(bp + k0 + bSel + 8));
    c = __builtin_amdgcn_wmma_f32_16x16x32_bf16(false, av, false, bv,
                                                (short)0, c, false, false);
  }

  const float bv = bias[colB];
  const int row0 = m0 + ((lane < 16) ? 0 : 8);
#pragma unroll
  for (int r = 0; r < 8; ++r)
    out[(size_t)(row0 + r) * EMB + colB] = c[r] + bv;
}

// ---------------------------------------------------------------------------
// Host-side launch.
// ---------------------------------------------------------------------------
extern "C" void kernel_launch(void* const* d_in, const int* in_sizes, int n_in,
                              void* d_out, int out_size, void* d_ws, size_t ws_size,
                              hipStream_t stream) {
  (void)in_sizes; (void)n_in; (void)out_size; (void)ws_size;

  const float* x  = (const float*)d_in[0];
  const float* wk = (const float*)d_in[1];   // note setup order: keys first
  const float* wq = (const float*)d_in[2];
  const float* wv = (const float*)d_in[3];
  const float* wu = (const float*)d_in[4];
  const float* bu = (const float*)d_in[5];
  float* out = (float*)d_out;

  unsigned short* ws = (unsigned short*)d_ws;
  size_t off = 0;
  unsigned short* xb   = ws + off; off += (size_t)ROWS * EMB;     // 8 MiB
  unsigned short* wt   = ws + off; off += (size_t)4 * EMB * HH;   // 2 MiB (q,k,v,u)
  unsigned short* q_ws = ws + off; off += (size_t)ROWS * HH;      // 8 MiB
  unsigned short* k_ws = ws + off; off += (size_t)ROWS * HH;      // 8 MiB
  unsigned short* vt_ws= ws + off; off += (size_t)ROWS * HH;      // 8 MiB
  unsigned short* attn = ws + off; off += (size_t)ROWS * HH;      // 8 MiB

  k_cvt_x<<<dim3(4096), dim3(256), 0, stream>>>((const float4*)x, xb);
  k_tw<<<dim3(1024, 4), dim3(256), 0, stream>>>(wq, wk, wv, wu, wt);
  k_qkv<<<dim3(ROWS / 16, 8, 3), dim3(128), 0, stream>>>(xb, wt, q_ws, k_ws, vt_ws);
  k_attn<<<dim3(NTOK / 16 / 4, HEADS, BS), dim3(128), 0, stream>>>(q_ws, k_ws, vt_ws, attn);
  k_unify<<<dim3(ROWS / 16, 8), dim3(128), 0, stream>>>(attn, wt + (size_t)3 * EMB * HH, bu, out);
}
